// InfiniAttention_38500086841985
// MI455X (gfx1250) — matrix operand order repaired and checked by reading, not compiled, not run on hardware
//
#include <hip/hip_runtime.h>
#include <hip/hip_bf16.h>

typedef __attribute__((ext_vector_type(16))) _Float16 v16h;
typedef __attribute__((ext_vector_type(8)))  _Float16 v8h;
typedef __attribute__((ext_vector_type(8)))  float    v8f;

#define B_  2
#define L_  2048
#define D_  2048
#define H_  16
#define HD  128
#define BL  4096      // B_*L_
#define HDH 2048      // H_*HD

// ---------------------------------------------------------------------------
// A/B fragment assembly: two 16-byte chunks -> v16h
// ---------------------------------------------------------------------------
__device__ __forceinline__ v16h frag16(const _Float16* p0, const _Float16* p1) {
    v8h a = *(const v8h*)p0;
    v8h b = *(const v8h*)p1;
    v16h r;
#pragma unroll
    for (int i = 0; i < 8; ++i) { r[i] = a[i]; r[i + 8] = b[i]; }
    return r;
}

// ---------------------------------------------------------------------------
// CDNA5 async global->LDS DMA (ASYNCcnt-tracked), per CDNA5 ISA + §15.18.
// LDS destination offset: low 32 bits of the flat shared-memory pointer
// (ISA: LDS aperture addresses truncate to the LDS offset).
// ---------------------------------------------------------------------------
__device__ __forceinline__ unsigned lds_off(const void* p) {
    return (unsigned)(unsigned long long)p;
}
__device__ __forceinline__ void async_copy_b128(unsigned ldsdst, const _Float16* gsrc) {
    asm volatile("global_load_async_to_lds_b128 %0, %1, off"
                 : : "v"(ldsdst), "v"(gsrc) : "memory");
}
__device__ __forceinline__ void wait_async0() {
    asm volatile("s_wait_asynccnt 0x0" ::: "memory");
}

// ---------------------------------------------------------------------------
// f32 -> f16 conversion
// ---------------------------------------------------------------------------
__global__ void cvt_f32_f16(const float* __restrict__ in, _Float16* __restrict__ out, int n) {
    int i = blockIdx.x * blockDim.x + threadIdx.x;
    if (i < n) out[i] = (_Float16)in[i];
}

// ---------------------------------------------------------------------------
// Tiled WMMA GEMM: C[MxN] = A[MxK] * B[KxN], f16 in, f32 accumulate.
// Block tile 64x128, K-step 32. 256 threads = 8 waves; wave computes 16x64.
// LDS tiles are filled by async global->LDS DMA (no VGPR round trip).
// ---------------------------------------------------------------------------
template<bool OUT_F32>
__global__ __launch_bounds__(256) void gemm16(const _Float16* __restrict__ A,
                                              const _Float16* __restrict__ B,
                                              void* __restrict__ C,
                                              int M, int N, int K) {
    __shared__ __align__(16) _Float16 As[64][40];    // 64x32 tile (+pad)
    __shared__ __align__(16) _Float16 Bs[32][136];   // 32x128 tile (+pad)

    const int tid  = threadIdx.x;
    const int lane = tid & 31;
    const int wave = tid >> 5;
    const int r    = lane & 15;
    const int hi   = lane >> 4;
    const int wm   = wave & 3;    // m-tile 0..3
    const int wn   = wave >> 2;   // n-half 0..1 (64 cols)
    const int m0   = blockIdx.y * 64;
    const int n0   = blockIdx.x * 128;

    v8f acc[4];
#pragma unroll
    for (int t = 0; t < 4; ++t)
#pragma unroll
        for (int i = 0; i < 8; ++i) acc[t][i] = 0.f;

    const int arow = tid >> 2, aseg = tid & 3;   // A: 64 rows x 4 segs of 8
    const int brow = tid >> 3, bseg = tid & 7;   // B: 32 rows x 8 segs of 16

    const unsigned la  = lds_off(&As[arow][aseg * 8]);
    const unsigned lb0 = lds_off(&Bs[brow][bseg * 16]);
    const unsigned lb1 = lds_off(&Bs[brow][bseg * 16 + 8]);

    for (int kk = 0; kk < K; kk += 32) {
        __syncthreads();   // previous tile fully consumed
        async_copy_b128(la,  &A[(size_t)(m0 + arow) * K + kk + aseg * 8]);
        async_copy_b128(lb0, &B[(size_t)(kk + brow) * N + n0 + bseg * 16]);
        async_copy_b128(lb1, &B[(size_t)(kk + brow) * N + n0 + bseg * 16 + 8]);
        if (kk + 32 < K) {   // prefetch next tiles -> global_prefetch_b8
            __builtin_prefetch(&A[(size_t)(m0 + arow) * K + kk + 32 + aseg * 8], 0, 0);
            __builtin_prefetch(&B[(size_t)(kk + 32 + brow) * N + n0 + bseg * 16], 0, 0);
        }
        wait_async0();       // this wave's DMA done
        __syncthreads();     // all waves' DMA visible

        v16h af = frag16(&As[wm * 16 + r][hi * 8], &As[wm * 16 + r][16 + hi * 8]);
#pragma unroll
        for (int t = 0; t < 4; ++t) {
            v16h bf;
#pragma unroll
            for (int s = 0; s < 16; ++s)
                bf[s] = Bs[(s & 7) + hi * 8 + (s >> 3) * 16][wn * 64 + t * 16 + r];
            acc[t] = __builtin_amdgcn_wmma_f32_16x16x32_f16(
                false, af, false, bf, (short)0, acc[t], false, false);
        }
    }

#pragma unroll
    for (int t = 0; t < 4; ++t)
#pragma unroll
        for (int i = 0; i < 8; ++i) {
            size_t m = m0 + wm * 16 + hi * 8 + i;
            size_t n = n0 + wn * 64 + t * 16 + r;
            float v = acc[t][i];
            if (OUT_F32) ((float*)C)[m * N + n] = v;
            else         ((_Float16*)C)[m * N + n] = (_Float16)v;
        }
}

// ---------------------------------------------------------------------------
// RoPE + elu(q)+1 + k-transpose. One thread per (b,h,l,d<64) pair.
// qkvh holds [q | k | v] as three 4096x2048 f16 matrices in (B,L,H*hd) order.
// ---------------------------------------------------------------------------
__global__ void rope_kernel(const _Float16* __restrict__ qkvh,
                            _Float16* __restrict__ qb,   // (B,H,L,hd) f16
                            _Float16* __restrict__ kT,   // (B,H,hd,L) f16
                            float* __restrict__ qf) {    // (B,H,L,hd) f32
    size_t tid = (size_t)blockIdx.x * blockDim.x + threadIdx.x;
    if (tid >= (size_t)B_ * H_ * L_ * 64) return;
    int d = tid & 63;
    int l = (int)(tid >> 6) & (L_ - 1);
    int h = (int)(tid >> 17) & (H_ - 1);
    int b = (int)(tid >> 21);

    const _Float16* qrow = qkvh + ((size_t)(b * L_ + l)) * HDH + h * HD;
    const _Float16* krow = qrow + (size_t)BL * HDH;
    float q1 = (float)qrow[d], q2 = (float)qrow[d + 64];
    float k1 = (float)krow[d], k2 = (float)krow[d + 64];
    // inv_freq = 10000^(-d/64) = exp(-d * ln(10000)/64)
    float inv = __expf(-(float)d * 0.14391156f);
    float ang = (float)l * inv;
    float c = __cosf(ang), s = __sinf(ang);
    float q1r = q1 * c - q2 * s, q2r = q1 * s + q2 * c;
    float k1r = k1 * c - k2 * s, k2r = k1 * s + k2 * c;

    size_t base = ((size_t)(b * H_ + h) * L_ + l) * HD;
    qb[base + d]      = (_Float16)q1r;
    qb[base + d + 64] = (_Float16)q2r;
    qf[base + d]      = q1r > 0.f ? q1r + 1.f : __expf(q1r);  // elu(x)+1
    qf[base + d + 64] = q2r > 0.f ? q2r + 1.f : __expf(q2r);
    size_t tb = (size_t)(b * H_ + h) * HD;
    kT[(tb + d) * L_ + l]        = (_Float16)k1r;
    kT[(tb + d + 64) * L_ + l]   = (_Float16)k2r;
}

// ---------------------------------------------------------------------------
// Memory path: comb[b,h,l,e] = sigmoid(gate_h) * (q_f . memory[:,e]) / (q_f . norm)
// One block per (b,h,l); 128 threads = one output element each.
// ---------------------------------------------------------------------------
__global__ __launch_bounds__(128) void mem_kernel(const float* __restrict__ qf,
                                                  const float* __restrict__ memory,
                                                  const float* __restrict__ norm,
                                                  const float* __restrict__ gate,
                                                  float* __restrict__ comb) {
    int e = threadIdx.x;
    int idx = blockIdx.x;
    int l = idx & (L_ - 1);
    int h = (idx >> 11) & (H_ - 1);
    int b = idx >> 15;
    __shared__ float qrow[HD];
    size_t base = ((size_t)(b * H_ + h) * L_ + l) * HD;
    qrow[e] = qf[base + e];
    __syncthreads();
    const float* mh = memory + (size_t)(b * H_ + h) * HD * HD;
    const float* nh = norm + (size_t)(b * H_ + h) * HD;
    float acc = 0.f, den = 0.f;
#pragma unroll 4
    for (int dd = 0; dd < HD; ++dd) {
        float q = qrow[dd];
        acc += q * mh[dd * HD + e];
        den += q * nh[dd];
    }
    float g = 1.f / (1.f + __expf(-gate[h]));
    comb[base + e] = g * acc / den;
}

// ---------------------------------------------------------------------------
// Flash attention + gating. 8 waves/WG; each wave owns a 16-row q tile,
// streams 32-key chunks: 8 WMMA for scores, online softmax, 8 WMMA for P*V.
// ---------------------------------------------------------------------------
__global__ __launch_bounds__(256) void attn_kernel(const _Float16* __restrict__ qb,
                                                   const _Float16* __restrict__ kT,
                                                   const _Float16* __restrict__ vh,
                                                   const float* __restrict__ comb_mem,
                                                   const float* __restrict__ gate,
                                                   _Float16* __restrict__ combh) {
    __shared__ __align__(16) _Float16 Pl[8][16][32];  // per-wave P tile
    const int tid  = threadIdx.x;
    const int lane = tid & 31;
    const int wave = tid >> 5;
    const int r    = lane & 15;
    const int hi   = lane >> 4;
    int blk = blockIdx.x;              // B*H*(L/128) = 512
    int qt  = blk & 15;
    int h   = (blk >> 4) & 15;
    int b   = blk >> 8;
    const int q0 = qt * 128 + wave * 16;
    const size_t bh = (size_t)(b * H_ + h);

    const _Float16* qbase = qb + bh * L_ * HD;
    const _Float16* ktb   = kT + bh * HD * L_;

    // Resident q A-fragments (16 rows x 128 d, 4 chunks of K=32)
    v16h aq[4];
#pragma unroll
    for (int c = 0; c < 4; ++c) {
        const _Float16* p = qbase + (size_t)(q0 + r) * HD + c * 32;
        aq[c] = frag16(p + hi * 8, p + 16 + hi * 8);
    }

    v8f accd[8];
#pragma unroll
    for (int t = 0; t < 8; ++t)
#pragma unroll
        for (int i = 0; i < 8; ++i) accd[t][i] = 0.f;
    float row_m[8], row_l[8];
#pragma unroll
    for (int i = 0; i < 8; ++i) { row_m[i] = -1e30f; row_l[i] = 0.f; }
    const float scale = 0.08838834764831845f;  // 1/sqrt(128)

    for (int k0 = 0; k0 < L_; k0 += 32) {
        v8f s0, s1;
#pragma unroll
        for (int i = 0; i < 8; ++i) { s0[i] = 0.f; s1[i] = 0.f; }
#pragma unroll
        for (int c = 0; c < 4; ++c) {
            v16h b0, b1;
#pragma unroll
            for (int s = 0; s < 16; ++s) {
                const _Float16* kp = &ktb[(size_t)(c * 32 + (s & 7) + hi * 8 + (s >> 3) * 16) * L_ + k0 + r];
                b0[s] = kp[0];
                b1[s] = kp[16];
            }
            s0 = __builtin_amdgcn_wmma_f32_16x16x32_f16(false, aq[c], false, b0,
                                                        (short)0, s0, false, false);
            s1 = __builtin_amdgcn_wmma_f32_16x16x32_f16(false, aq[c], false, b1,
                                                        (short)0, s1, false, false);
        }
        // Online softmax; reductions stay inside the 16-lane N-groups.
        float alpha[8];
#pragma unroll
        for (int i = 0; i < 8; ++i) {
            float x0 = s0[i] * scale, x1 = s1[i] * scale;
            float mx = fmaxf(x0, x1);
#pragma unroll
            for (int off = 8; off >= 1; off >>= 1) mx = fmaxf(mx, __shfl_xor(mx, off, 32));
            float nm = fmaxf(row_m[i], mx);
            alpha[i] = __expf(row_m[i] - nm);
            row_m[i] = nm;
            float p0 = __expf(x0 - nm), p1 = __expf(x1 - nm);
            float ps = p0 + p1;
#pragma unroll
            for (int off = 8; off >= 1; off >>= 1) ps += __shfl_xor(ps, off, 32);
            row_l[i] = row_l[i] * alpha[i] + ps;
            s0[i] = p0; s1[i] = p1;
        }
#pragma unroll
        for (int t = 0; t < 8; ++t)
#pragma unroll
            for (int i = 0; i < 8; ++i) accd[t][i] *= alpha[i];

        // C-layout -> A-layout for P via per-wave LDS tile
#pragma unroll
        for (int i = 0; i < 8; ++i) {
            Pl[wave][hi * 8 + i][r]      = (_Float16)s0[i];
            Pl[wave][hi * 8 + i][16 + r] = (_Float16)s1[i];
        }
        __syncthreads();
        v16h pa = frag16(&Pl[wave][r][hi * 8], &Pl[wave][r][16 + hi * 8]);
#pragma unroll
        for (int t = 0; t < 8; ++t) {
            v16h bv;
#pragma unroll
            for (int s = 0; s < 16; ++s)
                bv[s] = vh[((size_t)(b * L_ + k0 + (s & 7) + hi * 8 + (s >> 3) * 16)) * HDH + h * HD + t * 16 + r];
            accd[t] = __builtin_amdgcn_wmma_f32_16x16x32_f16(false, pa, false, bv,
                                                             (short)0, accd[t], false, false);
        }
        __syncthreads();
    }

    float g = 1.f / (1.f + __expf(-gate[h]));
    const float* cm = comb_mem + bh * L_ * HD;
#pragma unroll
    for (int t = 0; t < 8; ++t)
#pragma unroll
        for (int i = 0; i < 8; ++i) {
            int m = q0 + hi * 8 + i;
            int d = t * 16 + r;
            float o = (1.f - g) * (accd[t][i] / row_l[i]) + cm[(size_t)m * HD + d];
            combh[((size_t)(b * L_ + m)) * HDH + h * HD + d] = (_Float16)o;
        }
}

// ---------------------------------------------------------------------------
extern "C" void kernel_launch(void* const* d_in, const int* in_sizes, int n_in,
                              void* d_out, int out_size, void* d_ws, size_t ws_size,
                              hipStream_t stream) {
    const float* x    = (const float*)d_in[0];
    const float* Wq   = (const float*)d_in[1];
    const float* Wk   = (const float*)d_in[2];
    const float* Wv   = (const float*)d_in[3];
    const float* Wo   = (const float*)d_in[4];
    const float* gate = (const float*)d_in[5];
    const float* mem  = (const float*)d_in[6];
    const float* norm = (const float*)d_in[7];

    char* ws = (char*)d_ws;
    const size_t MiB = 1024 * 1024;
    _Float16* xh    = (_Float16*)(ws);               // 16 MiB
    _Float16* Wh    = (_Float16*)(ws + 16  * MiB);   // 32 MiB (Wq|Wk|Wv|Wo)
    _Float16* qkvh  = (_Float16*)(ws + 48  * MiB);   // 48 MiB (q|k|v)
    _Float16* qbh   = (_Float16*)(ws + 96  * MiB);   // 16 MiB
    _Float16* kTh   = (_Float16*)(ws + 112 * MiB);   // 16 MiB
    float*    qf    = (float*)   (ws + 128 * MiB);   // 32 MiB
    float*    combm = (float*)   (ws + 160 * MiB);   // 32 MiB
    _Float16* combh = (_Float16*)(ws + 192 * MiB);   // 16 MiB

    const int MAT = 2048 * 2048;     // one 2048x2048 matrix
    const int XSZ = 4096 * 2048;     // 4096x2048 activation

    cvt_f32_f16<<<(XSZ + 255) / 256, 256, 0, stream>>>(x,  xh, XSZ);
    cvt_f32_f16<<<(MAT + 255) / 256, 256, 0, stream>>>(Wq, Wh,                   MAT);
    cvt_f32_f16<<<(MAT + 255) / 256, 256, 0, stream>>>(Wk, Wh + (size_t)MAT,     MAT);
    cvt_f32_f16<<<(MAT + 255) / 256, 256, 0, stream>>>(Wv, Wh + (size_t)2 * MAT, MAT);
    cvt_f32_f16<<<(MAT + 255) / 256, 256, 0, stream>>>(Wo, Wh + (size_t)3 * MAT, MAT);

    dim3 gg(2048 / 128, 4096 / 64);  // (16, 64)
    gemm16<false><<<gg, 256, 0, stream>>>(xh, Wh,                   qkvh,                   4096, 2048, 2048);
    gemm16<false><<<gg, 256, 0, stream>>>(xh, Wh + (size_t)MAT,     qkvh + (size_t)XSZ,     4096, 2048, 2048);
    gemm16<false><<<gg, 256, 0, stream>>>(xh, Wh + (size_t)2 * MAT, qkvh + (size_t)2 * XSZ, 4096, 2048, 2048);

    size_t nrope = (size_t)B_ * H_ * L_ * 64;
    rope_kernel<<<(unsigned)((nrope + 255) / 256), 256, 0, stream>>>(qkvh, qbh, kTh, qf);

    mem_kernel<<<B_ * H_ * L_, 128, 0, stream>>>(qf, mem, norm, gate, combm);

    attn_kernel<<<B_ * H_ * (L_ / 128), 256, 0, stream>>>(qbh, kTh, qkvh + (size_t)2 * XSZ,
                                                          combm, gate, combh);

    gemm16<true><<<gg, 256, 0, stream>>>(combh, Wh + (size_t)3 * MAT, d_out, 4096, 2048, 2048);
}
